// AttentionBlock_24764781429183
// MI455X (gfx1250) — matrix operand order repaired and checked by reading, not compiled, not run on hardware
//
#include <hip/hip_runtime.h>
#include <hip/hip_bf16.h>
#include <math.h>

typedef _Float16 half_t;
typedef __attribute__((ext_vector_type(16))) _Float16 v16h;
typedef __attribute__((ext_vector_type(8)))  _Float16 v8h;
typedef __attribute__((ext_vector_type(8)))  float    v8f;

#define BB     4
#define CC     512
#define NN     4096
#define NGRP   32
#define CPG    16      // channels per group
#define GN_EPS 1e-5f

// ---------------------------------------------------------------------------
// WMMA helpers (CDNA5 wave32, 16x16x32 f16 -> f32)
// ---------------------------------------------------------------------------
__device__ __forceinline__ v8f wmma_f16(v16h a, v16h b, v8f c) {
  // (neg_a, A, neg_b, B, c_mod, C, reuse_a, reuse_b)
  return __builtin_amdgcn_wmma_f32_16x16x32_f16(false, a, false, b, (short)0, c,
                                                false, false);
}

// A fragment 16x32 (MxK), source row-major with leading dim `ld` (halves).
// lane L: row = row0 + L%16 ; halves 0..7 = K k0+(L/16)*8.., halves 8..15 = +16
__device__ __forceinline__ v16h load_a_frag(const half_t* base, int ld,
                                            int row0, int k0, int lane) {
  const half_t* p = base + (size_t)(row0 + (lane & 15)) * ld + k0 + ((lane >> 4) << 3);
  v8h lo = *(const v8h*)p;
  v8h hi = *(const v8h*)(p + 16);
  return __builtin_shufflevector(lo, hi, 0, 1, 2, 3, 4, 5, 6, 7,
                                 8, 9, 10, 11, 12, 13, 14, 15);
}

// B fragment 32x16 (KxN). Memory holds source[col][k] contiguous in k
// (leading dim ld). lane L: col = col0 + L%16 ; halves 0..7 = K k0+(L/16)*8.., halves 8..15 = +16
__device__ __forceinline__ v16h load_b_frag(const half_t* base, int ld,
                                            int col0, int k0, int lane) {
  const half_t* p = base + (size_t)(col0 + (lane & 15)) * ld + k0 + ((lane >> 4) << 3);
  v8h lo = *(const v8h*)p;
  v8h hi = *(const v8h*)(p + 16);
  return __builtin_shufflevector(lo, hi, 0, 1, 2, 3, 4, 5, 6, 7,
                                 8, 9, 10, 11, 12, 13, 14, 15);
}

// ---------------------------------------------------------------------------
// f32 -> f16 weight conversion
// ---------------------------------------------------------------------------
__global__ __launch_bounds__(256) void f32_to_f16(const float* __restrict__ src,
                                                  half_t* __restrict__ dst, int n) {
  int i = blockIdx.x * 256 + threadIdx.x;
  if (i < n) dst[i] = (half_t)src[i];
}

// ---------------------------------------------------------------------------
// GroupNorm stats: one block per (batch, group); 16 ch x 4096 px reduction
// ---------------------------------------------------------------------------
__global__ __launch_bounds__(256) void gn_stats(const float* __restrict__ x,
                                                float* __restrict__ stats) {
  int bg = blockIdx.x;                // 0 .. B*NGRP-1
  int bat = bg / NGRP, g = bg % NGRP;
  const float* xp = x + (size_t)bat * CC * NN + (size_t)g * CPG * NN;
  float s = 0.f, sq = 0.f;
  for (int i = threadIdx.x; i < CPG * NN; i += 256) {
    float v = xp[i];
    s += v; sq += v * v;
  }
  __shared__ float rs[256], rq[256];
  rs[threadIdx.x] = s; rq[threadIdx.x] = sq;
  __syncthreads();
  for (int off = 128; off > 0; off >>= 1) {
    if (threadIdx.x < off) {
      rs[threadIdx.x] += rs[threadIdx.x + off];
      rq[threadIdx.x] += rq[threadIdx.x + off];
    }
    __syncthreads();
  }
  if (threadIdx.x == 0) {
    float inv = 1.0f / (float)(CPG * NN);
    float mean = rs[0] * inv;
    float var  = rq[0] * inv - mean * mean;
    stats[bg * 2 + 0] = mean;
    stats[bg * 2 + 1] = rsqrtf(var + GN_EPS);
  }
}

// ---------------------------------------------------------------------------
// GroupNorm apply + transpose to token-major f16 T[B][N][C]
// ---------------------------------------------------------------------------
__global__ __launch_bounds__(256) void gn_apply(const float* __restrict__ x,
                                                const float* __restrict__ w,
                                                const float* __restrict__ bias,
                                                const float* __restrict__ stats,
                                                half_t* __restrict__ T) {
  int idx = blockIdx.x * 256 + threadIdx.x;   // over B*C*N (exact)
  int bat = idx / (CC * NN);
  int rem = idx % (CC * NN);
  int c = rem / NN;
  int n = rem % NN;
  int bg = bat * NGRP + (c / CPG);
  float mean = stats[bg * 2 + 0];
  float rstd = stats[bg * 2 + 1];
  float v = (x[idx] - mean) * rstd * w[c] + bias[c];
  T[(size_t)bat * NN * CC + (size_t)n * CC + c] = (half_t)v;
}

// ---------------------------------------------------------------------------
// Y[b,n,d] = (sum_c A[b,n,c] * W[d,c] + bias[d]) * scale
// 256 threads = 8 waves; block tile 128 rows x 64 cols; wave tile 32x32.
// transpose_out: 0 -> Y[b][n][d] (ld C), 1 -> Y[b][d][n] (ld N, for V^T)
// ---------------------------------------------------------------------------
__global__ __launch_bounds__(256) void gemm_nt(const half_t* __restrict__ A,
                                               const half_t* __restrict__ W,
                                               const float* __restrict__ bias,
                                               half_t* __restrict__ Y,
                                               float scale, int transpose_out) {
  int lane = threadIdx.x & 31;
  int wid  = threadIdx.x >> 5;
  int bat  = blockIdx.z;
  int m0 = blockIdx.x * 128 + (wid & 3) * 32;
  int d0 = blockIdx.y * 64 + (wid >> 2) * 32;
  const half_t* Ab = A + (size_t)bat * NN * CC;

  v8f acc[2][2] = {};
  for (int k0 = 0; k0 < CC; k0 += 32) {
    v16h a0 = load_a_frag(Ab, CC, m0, k0, lane);
    v16h a1 = load_a_frag(Ab, CC, m0 + 16, k0, lane);
    v16h b0 = load_b_frag(W, CC, d0, k0, lane);
    v16h b1 = load_b_frag(W, CC, d0 + 16, k0, lane);
    acc[0][0] = wmma_f16(a0, b0, acc[0][0]);
    acc[0][1] = wmma_f16(a0, b1, acc[0][1]);
    acc[1][0] = wmma_f16(a1, b0, acc[1][0]);
    acc[1][1] = wmma_f16(a1, b1, acc[1][1]);
  }

  int rbase = (lane >> 4) << 3;
  for (int j = 0; j < 2; ++j) {
    int col = d0 + j * 16 + (lane & 15);
    float bval = bias[col];
    for (int i = 0; i < 2; ++i) {
      int row0 = m0 + i * 16 + rbase;
      for (int r = 0; r < 8; ++r) {
        float y = (acc[i][j][r] + bval) * scale;
        if (transpose_out)
          Y[(size_t)bat * CC * NN + (size_t)col * NN + (row0 + r)] = (half_t)y;
        else
          Y[(size_t)bat * NN * CC + (size_t)(row0 + r) * CC + col] = (half_t)y;
      }
    }
  }
}

// ---------------------------------------------------------------------------
// Flash-style attention. One block = 16 query rows; 8 waves.
// Loop over 128-key blocks: wave w computes S tile for keys [kb+16w, +16),
// online softmax by 16 threads, then each wave owns 64 output channels.
// Softmax scale is pre-folded into Q.
// ---------------------------------------------------------------------------
__global__ __launch_bounds__(256) void attention_fa(const half_t* __restrict__ Q,
                                                    const half_t* __restrict__ K,
                                                    const half_t* __restrict__ Vt,
                                                    half_t* __restrict__ O) {
  int lane = threadIdx.x & 31;
  int wid  = threadIdx.x >> 5;
  int bat  = blockIdx.y;
  int nq0  = blockIdx.x * 16;

  __shared__ __align__(32) half_t Qs[16 * CC];   // 16 KB
  __shared__ __align__(32) float  Ss[16 * 128];  // 8 KB
  __shared__ __align__(32) half_t Ps[16 * 128];  // 4 KB
  __shared__ float mrun[16], lrun[16], corr[16];

  // stage the Q tile (16 x 512 halves) into LDS
  {
    const uint32_t* src = (const uint32_t*)(Q + (size_t)bat * NN * CC + (size_t)nq0 * CC);
    uint32_t* dst = (uint32_t*)Qs;
    for (int i = threadIdx.x; i < 16 * CC / 2; i += 256) dst[i] = src[i];
  }
  if (threadIdx.x < 16) { mrun[threadIdx.x] = -INFINITY; lrun[threadIdx.x] = 0.f; }
  __syncthreads();

  const half_t* Kb = K + (size_t)bat * NN * CC;
  const half_t* Vb = Vt + (size_t)bat * CC * NN;
  int rbase = (lane >> 4) << 3;

  v8f oacc[4] = {};   // this wave's 16 rows x 64 channels

  for (int kb = 0; kb < NN; kb += 128) {
    // ---- S = Q x K^T for this wave's 16 keys
    v8f s = {};
    int key0 = kb + wid * 16;
    for (int k0 = 0; k0 < CC; k0 += 32) {
      v16h a  = load_a_frag(Qs, CC, 0, k0, lane);
      v16h bf = load_b_frag(Kb, CC, key0, k0, lane);
      s = wmma_f16(a, bf, s);
    }
    int colb = wid * 16 + (lane & 15);
    for (int r = 0; r < 8; ++r) Ss[(rbase + r) * 128 + colb] = s[r];
    __syncthreads();

    // ---- online softmax (one thread per query row)
    if (threadIdx.x < 16) {
      int row = threadIdx.x;
      float mb = -INFINITY;
      for (int j = 0; j < 128; ++j) mb = fmaxf(mb, Ss[row * 128 + j]);
      float mnew = fmaxf(mrun[row], mb);
      float cfac = __expf(mrun[row] - mnew);
      float ssum = 0.f;
      for (int j = 0; j < 128; ++j) {
        float p = __expf(Ss[row * 128 + j] - mnew);
        Ps[row * 128 + j] = (half_t)p;
        ssum += p;
      }
      lrun[row] = lrun[row] * cfac + ssum;
      mrun[row] = mnew;
      corr[row] = cfac;
    }
    __syncthreads();

    // ---- rescale running O, then O += P x V over this key block
    float cr[8];
    for (int r = 0; r < 8; ++r) cr[r] = corr[rbase + r];
    for (int j = 0; j < 4; ++j)
      for (int r = 0; r < 8; ++r) oacc[j][r] *= cr[r];

    for (int k0 = 0; k0 < 128; k0 += 32) {
      v16h a = load_a_frag(Ps, 128, 0, k0, lane);
      for (int j = 0; j < 4; ++j) {
        v16h bf = load_b_frag(Vb, NN, wid * 64 + j * 16, kb + k0, lane);
        oacc[j] = wmma_f16(a, bf, oacc[j]);
      }
    }
    __syncthreads();   // protect Ss/Ps for next key block
  }

  // ---- finalize: divide by row sums, store O[b][n][d] as f16
  float li[8];
  for (int r = 0; r < 8; ++r) li[r] = 1.0f / lrun[rbase + r];
  half_t* Ob = O + (size_t)bat * NN * CC;
  for (int j = 0; j < 4; ++j) {
    int col = wid * 64 + j * 16 + (lane & 15);
    for (int r = 0; r < 8; ++r)
      Ob[(size_t)(nq0 + rbase + r) * CC + col] = (half_t)(oacc[j][r] * li[r]);
  }
}

// ---------------------------------------------------------------------------
// Output projection + residual: out[b][d][n] = x[b][d][n] +
//                               sum_c O[b][n][c]*Wp[d][c] + bp[d]
// ---------------------------------------------------------------------------
__global__ __launch_bounds__(256) void gemm_proj(const half_t* __restrict__ A,
                                                 const half_t* __restrict__ W,
                                                 const float* __restrict__ bias,
                                                 const float* __restrict__ x,
                                                 float* __restrict__ out) {
  int lane = threadIdx.x & 31;
  int wid  = threadIdx.x >> 5;
  int bat  = blockIdx.z;
  int m0 = blockIdx.x * 128 + (wid & 3) * 32;
  int d0 = blockIdx.y * 64 + (wid >> 2) * 32;
  const half_t* Ab = A + (size_t)bat * NN * CC;

  v8f acc[2][2] = {};
  for (int k0 = 0; k0 < CC; k0 += 32) {
    v16h a0 = load_a_frag(Ab, CC, m0, k0, lane);
    v16h a1 = load_a_frag(Ab, CC, m0 + 16, k0, lane);
    v16h b0 = load_b_frag(W, CC, d0, k0, lane);
    v16h b1 = load_b_frag(W, CC, d0 + 16, k0, lane);
    acc[0][0] = wmma_f16(a0, b0, acc[0][0]);
    acc[0][1] = wmma_f16(a0, b1, acc[0][1]);
    acc[1][0] = wmma_f16(a1, b0, acc[1][0]);
    acc[1][1] = wmma_f16(a1, b1, acc[1][1]);
  }

  int rbase = (lane >> 4) << 3;
  for (int j = 0; j < 2; ++j) {
    int col = d0 + j * 16 + (lane & 15);
    float bval = bias[col];
    for (int i = 0; i < 2; ++i) {
      int row0 = m0 + i * 16 + rbase;
      size_t base = (size_t)bat * CC * NN + (size_t)col * NN + row0;
      for (int r = 0; r < 8; ++r)
        out[base + r] = x[base + r] + acc[i][j][r] + bval;
    }
  }
}

// ---------------------------------------------------------------------------
// Host launcher
// ---------------------------------------------------------------------------
extern "C" void kernel_launch(void* const* d_in, const int* in_sizes, int n_in,
                              void* d_out, int out_size, void* d_ws, size_t ws_size,
                              hipStream_t stream) {
  (void)in_sizes; (void)n_in; (void)out_size; (void)ws_size;
  const float* x    = (const float*)d_in[0];
  const float* gn_w = (const float*)d_in[1];
  const float* gn_b = (const float*)d_in[2];
  const float* wq   = (const float*)d_in[3];
  const float* bq   = (const float*)d_in[4];
  const float* wk   = (const float*)d_in[5];
  const float* bk   = (const float*)d_in[6];
  const float* wv   = (const float*)d_in[7];
  const float* bv   = (const float*)d_in[8];
  const float* wp   = (const float*)d_in[9];
  const float* bp   = (const float*)d_in[10];
  float* out = (float*)d_out;

  // workspace layout (all offsets multiples of 1024 B)
  char* ws = (char*)d_ws;
  const size_t TOK_BYTES = (size_t)BB * NN * CC * sizeof(half_t);  // 16 MiB
  const size_t W_BYTES   = (size_t)CC * CC * sizeof(half_t);       // 512 KiB
  size_t off = 0;
  float*  stats = (float*)(ws + off);  off += 1024;
  half_t* T     = (half_t*)(ws + off); off += TOK_BYTES;
  half_t* Qb    = (half_t*)(ws + off); off += TOK_BYTES;
  half_t* Kb    = (half_t*)(ws + off); off += TOK_BYTES;
  half_t* Vtb   = (half_t*)(ws + off); off += TOK_BYTES;
  half_t* Ob    = (half_t*)(ws + off); off += TOK_BYTES;
  half_t* Wqh   = (half_t*)(ws + off); off += W_BYTES;
  half_t* Wkh   = (half_t*)(ws + off); off += W_BYTES;
  half_t* Wvh   = (half_t*)(ws + off); off += W_BYTES;
  half_t* Wph   = (half_t*)(ws + off); off += W_BYTES;

  const int wn = CC * CC;
  f32_to_f16<<<(wn + 255) / 256, 256, 0, stream>>>(wq, Wqh, wn);
  f32_to_f16<<<(wn + 255) / 256, 256, 0, stream>>>(wk, Wkh, wn);
  f32_to_f16<<<(wn + 255) / 256, 256, 0, stream>>>(wv, Wvh, wn);
  f32_to_f16<<<(wn + 255) / 256, 256, 0, stream>>>(wp, Wph, wn);

  gn_stats<<<BB * NGRP, 256, 0, stream>>>(x, stats);
  gn_apply<<<(BB * CC * NN) / 256, 256, 0, stream>>>(x, gn_w, gn_b, stats, T);

  dim3 gemm_grid(NN / 128, CC / 64, BB);
  const float qscale = 0.044194173824159216f;  // 1/sqrt(512), folded into Q
  gemm_nt<<<gemm_grid, 256, 0, stream>>>(T, Wqh, bq, Qb, qscale, 0);
  gemm_nt<<<gemm_grid, 256, 0, stream>>>(T, Wkh, bk, Kb, 1.0f, 0);
  gemm_nt<<<gemm_grid, 256, 0, stream>>>(T, Wvh, bv, Vtb, 1.0f, 1);  // V^T

  attention_fa<<<dim3(NN / 16, BB), 256, 0, stream>>>(Qb, Kb, Vtb, Ob);

  gemm_proj<<<gemm_grid, 256, 0, stream>>>(Ob, Wph, bp, x, out);
}
